// GPT2Attention_86388972192344
// MI455X (gfx1250) — matrix operand order repaired and checked by reading, not compiled, not run on hardware
//
#include <hip/hip_runtime.h>

// MI455X / gfx1250, wave32. GPT2 attention: B=2, S=2048, D=1024, H=16, hd=64.
// ~69 GFLOP vs ~100MB HBM traffic -> compute bound -> route all matmuls
// through v_wmma_f32_16x16x32_bf16 (bf16 operands, f32 accumulate).
// Attention K/Q tiles staged with GLOBAL_LOAD_ASYNC_TO_LDS_B128 (ASYNCcnt).

typedef __bf16 bf16_t;
typedef __attribute__((ext_vector_type(16))) __bf16 v16bf;
typedef __attribute__((ext_vector_type(8)))  float  v8f;

#define WMMA_BF16(A, B, C) \
  __builtin_amdgcn_wmma_f32_16x16x32_bf16(false, (A), false, (B), (short)0, (C), false, false)

static constexpr int Bc = 2, Sc = 2048, Dc = 1024, Hc = 16, HDc = 64;

// ---- CDNA5 async global->LDS copy (16B per lane), tracked by ASYNCcnt ----
__device__ inline void async_ld16(void* lds, const void* g) {
  uint32_t l = (uint32_t)(uintptr_t)lds;   // low 32 bits of flat LDS ptr = LDS offset
  asm volatile("global_load_async_to_lds_b128 %0, %1, off"
               :: "v"(l), "v"(g) : "memory");
}
__device__ inline void wait_async0() {
  asm volatile("s_wait_asynccnt 0x0" ::: "memory");
}

// ---- WMMA fragment loaders (CDNA5 16-bit layouts, CDNA5 ISA 7.12.2) ----
// A (16x32, row-major in LDS): lane m = lane&15, half = lane>>4
//   VGPR v holds K = (v&3)*2 + (v>>2)*16 + half*8  (pair K, K+1 contiguous)
__device__ inline v16bf frag_a(const bf16_t* base, int ld, int m, int hf, int k0) {
  v16bf a;
  const bf16_t* p = base + m * ld + k0 + hf * 8;
#pragma unroll
  for (int v = 0; v < 8; ++v) {
    int k = (v & 3) * 2 + (v >> 2) * 16;
    a[2 * v]     = p[k];
    a[2 * v + 1] = p[k + 1];
  }
  return a;
}
// B (32x16) staged transposed in LDS as Bt[n][k]: lane n = lane&15, half = lane>>4
//   VGPR v holds K = (v&3)*2 + (v>>2)*16 + half*8 (pair contiguous in Bt row)
__device__ inline v16bf frag_b(const bf16_t* base, int ld, int n, int hf, int k0) {
  v16bf b;
  const bf16_t* p = base + n * ld + k0 + hf * 8;
#pragma unroll
  for (int v = 0; v < 8; ++v) {
    int k = (v & 3) * 2 + (v >> 2) * 16;
    b[2 * v]     = p[k];
    b[2 * v + 1] = p[k + 1];
  }
  return b;
}

// =====================  Kernel 1: QKV GEMM (+bias, split heads)  =====================
// C[4096,3072] = X[4096,1024] @ Wattn[1024,3072] + b ; outputs q,k,v bf16 [B,H,S,hd]
__global__ __launch_bounds__(128)
void qkv_gemm(const float* __restrict__ x, const float* __restrict__ w,
              const float* __restrict__ bias,
              bf16_t* __restrict__ qo, bf16_t* __restrict__ ko, bf16_t* __restrict__ vo) {
  __shared__ __align__(16) bf16_t As[64 * 40];   // 64 rows x 32 k (pad->40)
  __shared__ __align__(16) bf16_t Bt[64 * 40];   // 64 n   x 32 k (transposed, pad->40)
  const int tn0 = blockIdx.x * 64;            // output col tile (0..3072)
  const int tm0 = blockIdx.y * 64;            // output row tile (0..4096)
  const int tid = threadIdx.x;
  const int wave = tid >> 5, lane = tid & 31;
  const int hf = lane >> 4, ln = lane & 15;

  v8f acc[4] = {};
  for (int k0 = 0; k0 < Dc; k0 += 32) {
    __syncthreads();
    // stage A tile 64x32 (f32 -> bf16)
#pragma unroll
    for (int e = 0; e < 4; ++e) {
      int idx = tid + e * 128;                // 512 quads
      int r = idx >> 3, c = (idx & 7) * 4;
      const float4 f = *(const float4*)(x + (size_t)(tm0 + r) * Dc + k0 + c);
      bf16_t* d = As + r * 40 + c;
      d[0] = (bf16_t)f.x; d[1] = (bf16_t)f.y; d[2] = (bf16_t)f.z; d[3] = (bf16_t)f.w;
    }
    // stage W tile 32x64 transposed into Bt[n][k]
#pragma unroll
    for (int e = 0; e < 4; ++e) {
      int idx = tid + e * 128;
      int kk = idx >> 4, j = (idx & 15) * 4;
      const float4 f = *(const float4*)(w + (size_t)(k0 + kk) * 3072 + tn0 + j);
      Bt[(j + 0) * 40 + kk] = (bf16_t)f.x; Bt[(j + 1) * 40 + kk] = (bf16_t)f.y;
      Bt[(j + 2) * 40 + kk] = (bf16_t)f.z; Bt[(j + 3) * 40 + kk] = (bf16_t)f.w;
    }
    __syncthreads();
    v16bf a = frag_a(As + wave * 16 * 40, 40, ln, hf, 0);
    v16bf b[4];
#pragma unroll
    for (int t = 0; t < 4; ++t) b[t] = frag_b(Bt + t * 16 * 40, 40, ln, hf, 0);
#pragma unroll
    for (int t = 0; t < 4; ++t) acc[t] = WMMA_BF16(a, b[t], acc[t]);
  }
  // epilogue: +bias, split into q/k/v [B,H,S,hd]
#pragma unroll
  for (int t = 0; t < 4; ++t) {
#pragma unroll
    for (int vv = 0; vv < 8; ++vv) {
      int m = vv + 8 * hf;
      int row = tm0 + wave * 16 + m;          // token index (0..4095)
      int col = tn0 + t * 16 + ln;            // 0..3071
      float val = acc[t][vv] + bias[col];
      int which = col >> 10;                  // 0=q 1=k 2=v
      int jj = col & 1023;
      int h = jj >> 6, c = jj & 63;
      int bb = row >> 11, s = row & 2047;
      bf16_t* dst = (which == 0) ? qo : (which == 1) ? ko : vo;
      dst[(((size_t)(bb * Hc + h)) * Sc + s) * HDc + c] = (bf16_t)val;
    }
  }
}

// =====================  Kernel 2: flash attention per (b,h, 64-row q-tile)  =====================
__global__ __launch_bounds__(128)
void attn_kernel(const bf16_t* __restrict__ q, const bf16_t* __restrict__ kmat,
                 const bf16_t* __restrict__ vmat, const float* __restrict__ mask,
                 bf16_t* __restrict__ ctx) {
  __shared__ __align__(16) bf16_t Qs[64 * 72];  // Q tile (rows q, cols hd)
  __shared__ __align__(16) bf16_t Ks[64 * 72];  // K tile row-major == Bt for Q*K^T
  __shared__ __align__(16) bf16_t Vt[64 * 72];  // V transposed: Vt[hd][krow] == Bt for P*V
  __shared__ float  Ss[4][16 * 65];             // per-wave score strip 16x64
  __shared__ __align__(16) bf16_t Ps[4][16 * 72];  // per-wave prob strip 16x64 (bf16)
  __shared__ float  mrow[4][16], lrow[4][16], arow[4][16];

  const int qt0 = blockIdx.x * 64;            // q-row base in S
  const int bh  = blockIdx.y;                 // 0..B*H-1
  const int bb  = bh >> 4;                    // batch (H=16)
  const int tid = threadIdx.x;
  const int wave = tid >> 5, lane = tid & 31;
  const int hf = lane >> 4, ln = lane & 15;

  const bf16_t* Qg = q    + ((size_t)bh * Sc + qt0) * HDc;
  const bf16_t* Kg = kmat + (size_t)bh * Sc * HDc;
  const bf16_t* Vg = vmat + (size_t)bh * Sc * HDc;
  const float*  mk = mask + (size_t)bb * Sc;

  // Q tile: async global->LDS (ASYNCcnt), 16B per lane per issue
#pragma unroll
  for (int e = 0; e < 4; ++e) {
    int idx = tid + e * 128;
    int r = idx >> 3, c = (idx & 7) * 8;
    async_ld16(&Qs[r * 72 + c], Qg + (size_t)r * 64 + c);
  }
  if (tid < 64) { mrow[tid >> 4][tid & 15] = -3.0e38f; lrow[tid >> 4][tid & 15] = 0.f; }

  v8f O[4] = {};
  for (int kt = 0; kt < Sc; kt += 64) {
    __syncthreads();
    // stage K tile (async straight copy) + V tile transposed (ds stores)
#pragma unroll
    for (int e = 0; e < 4; ++e) {
      int idx = tid + e * 128;
      int r = idx >> 3, c = (idx & 7) * 8;
      async_ld16(&Ks[r * 72 + c], Kg + (size_t)(kt + r) * 64 + c);
    }
    for (int e = 0; e < 32; ++e) {
      int idx = tid + e * 128;
      int r = idx >> 6, c = idx & 63;
      Vt[c * 72 + r] = Vg[(size_t)(kt + r) * 64 + c];
    }
    if (kt + 64 < Sc) {  // warm L2 for the next tiles
      __builtin_prefetch(Kg + (size_t)(kt + 64 + (tid >> 1)) * 64, 0, 1);
      __builtin_prefetch(Vg + (size_t)(kt + 64 + (tid >> 1)) * 64, 0, 1);
    }
    wait_async0();
    __syncthreads();

    // scores: S = Q @ K^T  (k-dim = hd = 64 -> two WMMA per 16x16 tile)
    v16bf a0 = frag_a(Qs + wave * 16 * 72, 72, ln, hf, 0);
    v16bf a1 = frag_a(Qs + wave * 16 * 72, 72, ln, hf, 32);
    v8f sc[4] = {};
    {
      v16bf kb[4];
#pragma unroll
      for (int t = 0; t < 4; ++t) kb[t] = frag_b(Ks + t * 16 * 72, 72, ln, hf, 0);
#pragma unroll
      for (int t = 0; t < 4; ++t) sc[t] = WMMA_BF16(a0, kb[t], sc[t]);
#pragma unroll
      for (int t = 0; t < 4; ++t) kb[t] = frag_b(Ks + t * 16 * 72, 72, ln, hf, 32);
#pragma unroll
      for (int t = 0; t < 4; ++t) sc[t] = WMMA_BF16(a1, kb[t], sc[t]);
    }
#pragma unroll
    for (int t = 0; t < 4; ++t)
#pragma unroll
      for (int vv = 0; vv < 8; ++vv) {
        int m = vv + 8 * hf;
        int col = t * 16 + ln;
        Ss[wave][m * 65 + col] = sc[t][vv] * 0.125f + mk[kt + col];
      }
    __syncthreads();

    // online softmax: lane r<16 of each wave owns row r of its strip
    if (lane < 16) {
      int r = lane;
      float mo = mrow[wave][r], lo = lrow[wave][r];
      float tm = mo;
      const float* srow = &Ss[wave][r * 65];
      for (int j = 0; j < 64; ++j) tm = fmaxf(tm, srow[j]);
      float al = __expf(mo - tm);
      float sum = 0.f;
      bf16_t* prow = &Ps[wave][r * 72];
      for (int j = 0; j < 64; ++j) {
        float p = __expf(srow[j] - tm);
        sum += p;
        prow[j] = (bf16_t)p;
      }
      mrow[wave][r] = tm;
      lrow[wave][r] = lo * al + sum;
      arow[wave][r] = al;
    }
    __syncthreads();

    // rescale running O, then O += P @ V
    float alv[8];
#pragma unroll
    for (int vv = 0; vv < 8; ++vv) alv[vv] = arow[wave][vv + 8 * hf];
#pragma unroll
    for (int t = 0; t < 4; ++t)
#pragma unroll
      for (int vv = 0; vv < 8; ++vv) O[t][vv] *= alv[vv];

    v16bf p0 = frag_a(Ps[wave], 72, ln, hf, 0);
    v16bf p1 = frag_a(Ps[wave], 72, ln, hf, 32);
    {
      v16bf vb[4];
#pragma unroll
      for (int t = 0; t < 4; ++t) vb[t] = frag_b(Vt + t * 16 * 72, 72, ln, hf, 0);
#pragma unroll
      for (int t = 0; t < 4; ++t) O[t] = WMMA_BF16(p0, vb[t], O[t]);
#pragma unroll
      for (int t = 0; t < 4; ++t) vb[t] = frag_b(Vt + t * 16 * 72, 72, ln, hf, 32);
#pragma unroll
      for (int t = 0; t < 4; ++t) O[t] = WMMA_BF16(p1, vb[t], O[t]);
    }
  }

  // epilogue: O /= l, store ctx bf16 [B,H,S,hd]
  __syncthreads();
  float linv[8];
#pragma unroll
  for (int vv = 0; vv < 8; ++vv) linv[vv] = 1.0f / lrow[wave][vv + 8 * hf];
  bf16_t* Cg = ctx + ((size_t)bh * Sc + qt0 + wave * 16) * HDc;
#pragma unroll
  for (int t = 0; t < 4; ++t)
#pragma unroll
    for (int vv = 0; vv < 8; ++vv) {
      int m = vv + 8 * hf;
      Cg[(size_t)m * 64 + t * 16 + ln] = (bf16_t)(O[t][vv] * linv[vv]);
    }
}

// =====================  Kernel 3: merge heads + out projection  =====================
// out[4096,1024] = ctx(merged)[4096,1024] @ Wproj[1024,1024] + b  (f32 out)
__global__ __launch_bounds__(128)
void proj_gemm(const bf16_t* __restrict__ ctx, const float* __restrict__ w,
               const float* __restrict__ bias, float* __restrict__ out) {
  __shared__ __align__(16) bf16_t As[64 * 40];
  __shared__ __align__(16) bf16_t Bt[64 * 40];
  const int tn0 = blockIdx.x * 64;
  const int tm0 = blockIdx.y * 64;
  const int tid = threadIdx.x;
  const int wave = tid >> 5, lane = tid & 31;
  const int hf = lane >> 4, ln = lane & 15;

  v8f acc[4] = {};
  for (int k0 = 0; k0 < Dc; k0 += 32) {
    __syncthreads();
    // stage A from ctx [B,H,S,hd] (merge heads via addressing)
    for (int e = 0; e < 16; ++e) {
      int idx = tid + e * 128;               // 2048 elements
      int r = idx >> 5, c = idx & 31;
      int row = tm0 + r, col = k0 + c;
      int bb = row >> 11, s = row & 2047, h = col >> 6, cc = col & 63;
      As[r * 40 + c] = ctx[(((size_t)(bb * Hc + h)) * Sc + s) * HDc + cc];
    }
    // stage Wproj tile 32x64 transposed
#pragma unroll
    for (int e = 0; e < 4; ++e) {
      int idx = tid + e * 128;
      int kk = idx >> 4, j = (idx & 15) * 4;
      const float4 f = *(const float4*)(w + (size_t)(k0 + kk) * Dc + tn0 + j);
      Bt[(j + 0) * 40 + kk] = (bf16_t)f.x; Bt[(j + 1) * 40 + kk] = (bf16_t)f.y;
      Bt[(j + 2) * 40 + kk] = (bf16_t)f.z; Bt[(j + 3) * 40 + kk] = (bf16_t)f.w;
    }
    __syncthreads();
    v16bf a = frag_a(As + wave * 16 * 40, 40, ln, hf, 0);
    v16bf b[4];
#pragma unroll
    for (int t = 0; t < 4; ++t) b[t] = frag_b(Bt + t * 16 * 40, 40, ln, hf, 0);
#pragma unroll
    for (int t = 0; t < 4; ++t) acc[t] = WMMA_BF16(a, b[t], acc[t]);
  }
#pragma unroll
  for (int t = 0; t < 4; ++t)
#pragma unroll
    for (int vv = 0; vv < 8; ++vv) {
      int row = tm0 + wave * 16 + vv + 8 * hf;
      int col = tn0 + t * 16 + ln;
      out[(size_t)row * Dc + col] = acc[t][vv] + bias[col];
    }
}

extern "C" void kernel_launch(void* const* d_in, const int* in_sizes, int n_in,
                              void* d_out, int out_size, void* d_ws, size_t ws_size,
                              hipStream_t stream) {
  const float* x    = (const float*)d_in[0];   // hidden_states [B,S,D]
  const float* mask = (const float*)d_in[1];   // attention_mask [B,1,1,S]
  const float* wa   = (const float*)d_in[2];   // w_attn [D,3D]
  const float* ba   = (const float*)d_in[3];   // b_attn [3D]
  const float* wp   = (const float*)d_in[4];   // w_proj [D,D]
  const float* bp   = (const float*)d_in[5];   // b_proj [D]
  float* out = (float*)d_out;

  bf16_t* ws  = (bf16_t*)d_ws;
  const size_t N = (size_t)Bc * Sc * Dc;       // 4M elements
  bf16_t* q   = ws;
  bf16_t* k   = ws + N;
  bf16_t* v   = ws + 2 * N;
  bf16_t* ctx = ws + 3 * N;                    // total 32MB of ws

  qkv_gemm  <<<dim3(48, 64), 128, 0, stream>>>(x, wa, ba, q, k, v);
  attn_kernel<<<dim3(32, 32), 128, 0, stream>>>(q, k, v, mask, ctx);
  proj_gemm <<<dim3(16, 64), 128, 0, stream>>>(ctx, wp, bp, out);
}
